// FourierLayer_81922206204258
// MI455X (gfx1250) — matrix operand order repaired and checked by reading, not compiled, not run on hardware
//
#include <hip/hip_runtime.h>
#include <stdint.h>

typedef _Float16 f16;
typedef __attribute__((ext_vector_type(16))) _Float16 v16h;
typedef __attribute__((ext_vector_type(8)))  float    v8f;

#define LPB      8      // lines per block
#define U_PITCH  264    // 256 + 8 pad: rows 16B-aligned, conflict-free b128 frag loads
#define G_PITCH  72     // 64 + 8 pad
#define H_PITCH  72     // 64 + 8 pad

// Dynamic LDS layout (bytes):
//   [0,       131072) : f32 staging, double buffered (2 x 64KB)
//   [131072,  164864) : lu  [64 c][U_PITCH] f16   (transposed line tile)
//   [164864,  238592) : lg  [32 k][16 rows][G_PITCH] f16 (rows = 2*l + {re,im})
//   [238592,  312320) : lh  [LPB l][64 o][H_PITCH] f16  (kk = k | 32+k)
#define SMEM_BYTES 312320

union ABFrag { v16h v; uint4 q[2]; };

// A-matrix f16 16x32 fragment (CDNA5 layout): lane L: row m = m0 + (L&15);
// halfs = [k0+8h .. +7] and [k0+16+8h .. +7], h = L>>4. Two 16B loads.
__device__ __forceinline__ v16h load_a_frag(const f16* __restrict__ base, int pitch,
                                            int m0, int k0, int lane) {
  ABFrag r;
  const f16* row = base + (m0 + (lane & 15)) * pitch + k0 + ((lane >> 4) << 3);
  r.q[0] = *(const uint4*)(row);
  r.q[1] = *(const uint4*)(row + 16);
  return r.v;
}

// B-matrix f16 32x16 fragment from [n][K]-major storage: lane L: col n = n0+(L&15);
// halfs = [k0+8h .. +7] and [k0+16+8h .. +7]. Two 16B loads.
__device__ __forceinline__ v16h load_b_frag(const f16* __restrict__ base, int pitch,
                                            int n0, int k0, int lane) {
  ABFrag r;
  const f16* row = base + (n0 + (lane & 15)) * pitch + k0 + ((lane >> 4) << 3);
  r.q[0] = *(const uint4*)(row);
  r.q[1] = *(const uint4*)(row + 16);
  return r.v;
}

// ---- init kernels ----------------------------------------------------------

// bF[f][p], f<32: cos(2*pi*f*p/256)/16 ; f>=32: -sin(2*pi*(f-32)*p/256)/16
// bI[p][kk], kk<32: c_k*cos/16 ; kk>=32: -c_k*sin/16  (c_0=1, else 2)
__global__ void build_basis_kernel(f16* __restrict__ bF, f16* __restrict__ bI) {
  int idx = blockIdx.x * blockDim.x + threadIdx.x;
  if (idx >= 64 * 256) return;
  int f = idx >> 8, p = idx & 255;
  int k = f & 31;
  float ang = 6.28318530717958647692f * (float)((k * p) & 255) * (1.0f / 256.0f);
  float cv = cosf(ang), sv = sinf(ang);
  float base = ((f < 32) ? cv : -sv) * 0.0625f;
  bF[f * 256 + p] = (f16)base;
  float ck = (k == 0) ? 1.0f : 2.0f;
  bI[p * 64 + f] = (f16)(base * ck);
}

// wt[(k*128 + ri*64 + o)*64 + c] = w[c][o][k][ri]   (w shape [64,64,32,2] f32)
__global__ void build_wt_kernel(const float* __restrict__ w, f16* __restrict__ wt) {
  int idx = blockIdx.x * blockDim.x + threadIdx.x;
  if (idx >= 32 * 128 * 64) return;
  int c = idx & 63;
  int n = (idx >> 6) & 127;
  int k = idx >> 13;
  int ri = n >> 6, o = n & 63;
  wt[idx] = (f16)w[(((c * 64) + o) * 32 + k) * 2 + ri];
}

// ---- async helpers ---------------------------------------------------------

// Issue async global->LDS copy of one raw f32 line tile (64 p x 64 c) into the
// staging buffer at LDS byte offset stgByteOff. 1024 16B chunks; each thread
// owns 4. Chunk q: p = q>>4, c4 = (q&15)*4.
__device__ __forceinline__ void async_load_tile(const float* __restrict__ src,
                                                long long posStride,
                                                uint32_t stgByteOff, int tid) {
  #pragma unroll
  for (int i = 0; i < 4; ++i) {
    int q = tid + 256 * i;
    int p = q >> 4, c4 = (q & 15) << 2;
    unsigned long long ga =
        (unsigned long long)(uintptr_t)(src + (long long)p * posStride + c4);
    uint32_t loff = stgByteOff + (uint32_t)q * 16u;
    asm volatile("global_load_async_to_lds_b128 %0, %1, off"
                 :: "v"(loff), "v"(ga) : "memory");
  }
}

__device__ __forceinline__ void wait_async() {
  asm volatile("s_wait_asynccnt 0x0" ::: "memory");
}

// stg (f32 [p][c]) -> lu (f16 [c][U_PITCH]) transpose + convert
__device__ __forceinline__ void transpose_tile(const float* __restrict__ stg,
                                               f16* __restrict__ lu, int tid) {
  #pragma unroll
  for (int i = 0; i < 4; ++i) {
    int q = tid + 256 * i;
    int p = q >> 4, c4 = (q & 15) << 2;
    float4 vv = *(const float4*)(stg + (size_t)q * 4);
    lu[(c4 + 0) * U_PITCH + p] = (f16)vv.x;
    lu[(c4 + 1) * U_PITCH + p] = (f16)vv.y;
    lu[(c4 + 2) * U_PITCH + p] = (f16)vv.z;
    lu[(c4 + 3) * U_PITCH + p] = (f16)vv.w;
  }
}

// ---- main pass: one spectral branch ---------------------------------------
// Lines: (b, r), base = b*B_STRIDE + r*lineStride; element (c,p) at +p*posStride + c.
// Branch Y: lineStride=16384 (x), posStride=64    (p = y), weights wt1
// Branch X: lineStride=64    (y), posStride=16384 (p = x), weights wt2

__global__ __launch_bounds__(256) void fno_spectral_pass(
    const float* __restrict__ x, float* __restrict__ out,
    const f16* __restrict__ bF, const f16* __restrict__ bI,
    const f16* __restrict__ wt,
    long long lineStride, long long posStride, int accumulate)
{
  extern __shared__ __align__(16) char smem[];
  f16* lu = (f16*)(smem + 131072);
  f16* lg = lu + 64 * U_PITCH;
  f16* lh = lg + 32 * 16 * G_PITCH;

  const int tid  = threadIdx.x;
  const int lane = tid & 31;
  const int wave = tid >> 5;
  const long long line0 = (long long)blockIdx.x * LPB;
  const long long batchStride = 256LL * 256LL * 64LL;

  // prologue: start streaming line 0 into staging buffer 0
  {
    const float* src0 = x + (line0 >> 8) * batchStride + (line0 & 255) * lineStride;
    async_load_tile(src0, posStride, 0u, tid);
  }

  // ---------------- Stage 1: per-line truncated forward DFT (WMMA) ---------
  for (int l = 0; l < LPB; ++l) {
    const uint32_t curOff = (l & 1) ? 65536u : 0u;
    wait_async();            // this wave's pending tile (line l) has landed in LDS
    __syncthreads();         // all waves' chunks visible; lu free (prev GEMM1 done)
    transpose_tile((const float*)(smem + curOff), lu, tid);
    if (l + 1 < LPB) {       // overlap: stream line l+1 while we compute line l
      long long nl = line0 + l + 1;
      const float* srcN = x + (nl >> 8) * batchStride + (nl & 255) * lineStride;
      async_load_tile(srcN, posStride, curOff ^ 65536u, tid);
    }
    __syncthreads();         // lu ready

    // D[c,f] = U[c,p] * F[f,p]^T ; wave w: c-tile w>>1 (A shared), f-tiles 2(w&1)+{0,1}
    const int ci  = (wave >> 1) * 16;
    const int f0a = ((wave & 1) * 2) * 16;
    const int f0b = f0a + 16;
    v8f acc0 = {}; v8f acc1 = {};
    #pragma unroll
    for (int ks = 0; ks < 8; ++ks) {
      v16h a  = load_a_frag(lu, U_PITCH, ci, ks * 32, lane);
      v16h b0 = load_b_frag(bF, 256, f0a, ks * 32, lane);
      v16h b1 = load_b_frag(bF, 256, f0b, ks * 32, lane);
      acc0 = __builtin_amdgcn_wmma_f32_16x16x32_f16(false, a, false, b0, (short)0, acc0, false, false);
      acc1 = __builtin_amdgcn_wmma_f32_16x16x32_f16(false, a, false, b1, (short)0, acc1, false, false);
    }
    // pack 8 f32 -> 8 f16 and store one b128 per tile: lg[(f&31)*16 + 2l + (f>>5)][c..c+7]
    const int co = ci + ((lane >> 4) << 3);
    {
      union { uint4 q; f16 h[8]; } g;
      int f = f0a + (lane & 15);
      int row = (f & 31) * 16 + 2 * l + (f >> 5);
      #pragma unroll
      for (int r = 0; r < 8; ++r) g.h[r] = (f16)acc0[r];
      *(uint4*)(lg + row * G_PITCH + co) = g.q;
      f = f0b + (lane & 15);
      row = (f & 31) * 16 + 2 * l + (f >> 5);
      #pragma unroll
      for (int r = 0; r < 8; ++r) g.h[r] = (f16)acc1[r];
      *(uint4*)(lg + row * G_PITCH + co) = g.q;
    }
  }
  __syncthreads();

  // ---------------- Stage 2: per-mode complex channel mix (WMMA) -----------
  for (int t = 0; t < 4; ++t) {
    int k = wave * 4 + t;                        // each wave owns 4 modes
    const f16* wk = wt + (size_t)k * 128 * 64;
    const f16* gk = lg + k * 16 * G_PITCH;
    v8f zero = {};
    v8f P[8];
    #pragma unroll
    for (int j = 0; j < 8; ++j) P[j] = zero;
    #pragma unroll
    for (int ks = 0; ks < 2; ++ks) {
      v16h a = load_a_frag(gk, G_PITCH, 0, ks * 32, lane);
      #pragma unroll
      for (int j = 0; j < 8; ++j) {              // n-tiles: Wre (0..3) | Wim (4..7)
        v16h b = load_b_frag(wk, 64, j * 16, ks * 32, lane);
        P[j] = __builtin_amdgcn_wmma_f32_16x16x32_f16(false, a, false, b, (short)0, P[j], false, false);
      }
    }
    // Hre = Gre*Wre - Gim*Wim ; Him = Gre*Wim + Gim*Wre (rows 2l / 2l+1 interleaved)
    const int l0 = (lane >> 4) << 2;
    const int oo = lane & 15;
    #pragma unroll
    for (int j = 0; j < 4; ++j) {
      #pragma unroll
      for (int i = 0; i < 4; ++i) {
        float hre = P[j][2 * i]     - P[j + 4][2 * i + 1];
        float him = P[j + 4][2 * i] + P[j][2 * i + 1];
        int basei = ((l0 + i) * 64 + j * 16 + oo) * H_PITCH;
        lh[basei + k]      = (f16)hre;
        lh[basei + 32 + k] = (f16)him;
      }
    }
  }
  __syncthreads();

  // ---------------- Stage 3: per-line inverse DFT (WMMA) + vector store ----
  for (int l = 0; l < LPB; ++l) {
    long long line = line0 + l;
    float* dst = out + (line >> 8) * batchStride + (line & 255) * lineStride;
    const f16* Hl = lh + l * 64 * H_PITCH;

    // A[o,kk] = H^T (from lh), hoisted across both p-tiles of this wave
    v16h A[4][2];
    #pragma unroll
    for (int oj = 0; oj < 4; ++oj) {
      A[oj][0] = load_a_frag(Hl, H_PITCH, oj * 16, 0,  lane);
      A[oj][1] = load_a_frag(Hl, H_PITCH, oj * 16, 32, lane);
    }
    #pragma unroll
    for (int pt = 0; pt < 2; ++pt) {
      int pj = wave * 2 + pt;                    // 16 p-tiles over 8 waves
      v16h b0 = load_b_frag(bI, 64, pj * 16, 0,  lane);
      v16h b1 = load_b_frag(bI, 64, pj * 16, 32, lane);
      int p = pj * 16 + (lane & 15);
      float* colbase = dst + (long long)p * posStride + ((lane >> 4) << 3);
      #pragma unroll
      for (int oj = 0; oj < 4; ++oj) {
        v8f acc = {};
        acc = __builtin_amdgcn_wmma_f32_16x16x32_f16(false, A[oj][0], false, b0, (short)0, acc, false, false);
        acc = __builtin_amdgcn_wmma_f32_16x16x32_f16(false, A[oj][1], false, b1, (short)0, acc, false, false);
        float* pd = colbase + oj * 16;           // 8 consecutive o per lane
        float4 lo, hi;
        lo.x = acc[0]; lo.y = acc[1]; lo.z = acc[2]; lo.w = acc[3];
        hi.x = acc[4]; hi.y = acc[5]; hi.z = acc[6]; hi.w = acc[7];
        if (accumulate) {
          float4 olo = *(const float4*)pd;
          float4 ohi = *(const float4*)(pd + 4);
          lo.x += olo.x; lo.y += olo.y; lo.z += olo.z; lo.w += olo.w;
          hi.x += ohi.x; hi.y += ohi.y; hi.z += ohi.z; hi.w += ohi.w;
        }
        *(float4*)pd = lo;
        *(float4*)(pd + 4) = hi;
      }
    }
  }
}

// ---- host launch -----------------------------------------------------------

extern "C" void kernel_launch(void* const* d_in, const int* in_sizes, int n_in,
                              void* d_out, int out_size, void* d_ws, size_t ws_size,
                              hipStream_t stream) {
  const float* x  = (const float*)d_in[0];
  const float* w1 = (const float*)d_in[1];
  const float* w2 = (const float*)d_in[2];
  // d_in[3], d_in[4]: modes_x / modes_y scalars == 32 (compile-time constant here)
  float* out = (float*)d_out;

  f16* ws  = (f16*)d_ws;
  f16* bF  = ws;                       // 64*256 halfs
  f16* bI  = bF + 64 * 256;            // 256*64 halfs
  f16* wt1 = bI + 256 * 64;            // 32*128*64 halfs
  f16* wt2 = wt1 + 32 * 128 * 64;      // 32*128*64 halfs   (total ~1.1 MB)

  build_basis_kernel<<<64, 256, 0, stream>>>(bF, bI);
  const int wtN = 32 * 128 * 64;
  build_wt_kernel<<<(wtN + 255) / 256, 256, 0, stream>>>(w1, wt1);
  build_wt_kernel<<<(wtN + 255) / 256, 256, 0, stream>>>(w2, wt2);

  (void)hipFuncSetAttribute((const void*)fno_spectral_pass,
                            hipFuncAttributeMaxDynamicSharedMemorySize, SMEM_BYTES);

  const int nLines = 16 * 256;                // (b, r) pairs per branch
  dim3 grid(nLines / LPB);                    // 512 blocks
  // Branch along Y (w1): lines (b, x); positions stride 64
  fno_spectral_pass<<<grid, 256, SMEM_BYTES, stream>>>(x, out, bF, bI, wt1, 16384LL, 64LL, 0);
  // Branch along X (w2): lines (b, y); positions stride 16384; accumulate
  fno_spectral_pass<<<grid, 256, SMEM_BYTES, stream>>>(x, out, bF, bI, wt2, 64LL, 16384LL, 1);
}
